// CartesianPlaneNonSirenEmbeddingNetwork_38036230373700
// MI455X (gfx1250) — matrix operand order repaired and checked by reading, not compiled, not run on hardware
//
#include <hip/hip_runtime.h>

// ---------------------------------------------------------------------------
// Triplane (CartesianPlaneNonSiren) embedding + tiny MLP, fused for gfx1250.
// All three MLP layers run on v_wmma_f32_16x16x32_f16; bilinear sampling runs
// on packed-f16 VALU (v_pk_fma_f16). Planes + weights converted to f16 once
// into d_ws by prep kernels; main kernel is one pass over the M points.
// ---------------------------------------------------------------------------

typedef _Float16 h8   __attribute__((ext_vector_type(8)));
typedef _Float16 v16h __attribute__((ext_vector_type(16)));
typedef float    v8f  __attribute__((ext_vector_type(8)));

union H16 { v16h v; h8 h[2]; };

#define WMMA_F16(a, b, c) \
  __builtin_amdgcn_wmma_f32_16x16x32_f16(false, (a), false, (b), (short)0, (c), false, false)

// A-matrix fragment (16x32 f16): lane l owns row (l&15).
// ISA layout: halves 0..7 = K = sub*8 .. sub*8+7 ; halves 8..15 = K = 16+sub*8..
// With row-major [row][k] f16 storage that is two contiguous b128 loads.
__device__ __forceinline__ v16h load_fragA(const _Float16* row, int ko, int sub) {
  H16 u;
  u.h[0] = *(const h8*)(row + ko + sub * 8);
  u.h[1] = *(const h8*)(row + ko + 16 + sub * 8);
  return u.v;
}

// B-matrix fragment (32x16 f16): lane l owns column (l&15).
// Same K order as A: halves 0..7 = K = sub*8 .. sub*8+7 ; halves 8..15 = K = 16+sub*8..
// -> two b128 loads of the [n][k] row.
__device__ __forceinline__ v16h load_fragB(const _Float16* row, int ko, int sub) {
  H16 u;
  u.h[0] = *(const h8*)(row + ko + sub * 8);
  u.h[1] = *(const h8*)(row + ko + 16 + sub * 8);
  return u.v;
}

// grid_sample(align_corners=True, padding_mode='zeros') corner setup for 32x32.
__device__ __forceinline__ void plane_setup(float gx, float gy, int* addr, _Float16* wt) {
  float fx = (gx + 1.0f) * 15.5f;
  float fy = (gy + 1.0f) * 15.5f;
  float x0f = floorf(fx), y0f = floorf(fy);
  float wx1 = fx - x0f, wy1 = fy - y0f;
  float wx0 = 1.0f - wx1, wy0 = 1.0f - wy1;
  int x0 = (int)x0f, y0 = (int)y0f;
  int x1 = x0 + 1, y1 = y0 + 1;
  float vx0 = (x0 >= 0 && x0 < 32) ? 1.0f : 0.0f;
  float vx1 = (x1 >= 0 && x1 < 32) ? 1.0f : 0.0f;
  float vy0 = (y0 >= 0 && y0 < 32) ? 1.0f : 0.0f;
  float vy1 = (y1 >= 0 && y1 < 32) ? 1.0f : 0.0f;
  int xc0 = min(max(x0, 0), 31), xc1 = min(max(x1, 0), 31);
  int yc0 = min(max(y0, 0), 31), yc1 = min(max(y1, 0), 31);
  wt[0] = (_Float16)(wx0 * wy0 * vx0 * vy0);
  wt[1] = (_Float16)(wx1 * wy0 * vx1 * vy0);
  wt[2] = (_Float16)(wx0 * wy1 * vx0 * vy1);
  wt[3] = (_Float16)(wx1 * wy1 * vx1 * vy1);
  addr[0] = (yc0 * 32 + xc0) * 64;
  addr[1] = (yc0 * 32 + xc1) * 64;
  addr[2] = (yc1 * 32 + xc0) * 64;
  addr[3] = (yc1 * 32 + xc1) * 64;
}

// Bilinear-accumulate 16 channels of one plane in packed f16 (v_pk_fma_f16).
__device__ __forceinline__ void sample16(const _Float16* __restrict__ pl,
                                         const int* addr, const _Float16* wt,
                                         int c0, h8& lo, h8& hi) {
  lo = h8{};
  hi = h8{};
#pragma unroll
  for (int k = 0; k < 4; ++k) {
    h8 v0 = *(const h8*)(pl + addr[k] + c0);
    h8 v1 = *(const h8*)(pl + addr[k] + c0 + 8);
    _Float16 wk = wt[k];
    lo += v0 * wk;   // 4x v_pk_fma_f16
    hi += v1 * wk;
  }
}

// --------------------------- prep kernels ----------------------------------

// (C,32,32) f32 -> (y,x,C) f16 for all 3 planes. 196608 threads.
__global__ void prep_planes(const float* __restrict__ pxy,
                            const float* __restrict__ pyz,
                            const float* __restrict__ pxz,
                            _Float16* __restrict__ dst) {
  int id = blockIdx.x * 256 + threadIdx.x;
  if (id >= 3 * 64 * 1024) return;
  int p = id >> 16;
  int rem = id & 65535;
  int c = rem >> 10;
  int yx = rem & 1023;
  const float* src = (p == 0) ? pxy : (p == 1) ? pyz : pxz;
  dst[p * 65536 + yx * 64 + c] = (_Float16)src[c * 1024 + yx];
}

// w1 (128x64), w2 (128x128) f32 -> f16 (row-major [o][c] == B's [n][k]),
// and w3 (1x128) -> zero-padded (16x128) f16 B matrix (rows 1..15 = 0).
__global__ void prep_weights(const float* __restrict__ w1,
                             const float* __restrict__ w2,
                             const float* __restrict__ w3,
                             _Float16* __restrict__ w1h,
                             _Float16* __restrict__ w2h,
                             _Float16* __restrict__ w3p) {
  int id = blockIdx.x * 256 + threadIdx.x;
  if (id < 8192) {
    w1h[id] = (_Float16)w1[id];
  } else if (id < 8192 + 16384) {
    int j = id - 8192;
    w2h[j] = (_Float16)w2[j];
  } else if (id < 8192 + 16384 + 2048) {
    int j = id - (8192 + 16384);
    int row = j >> 7, col = j & 127;
    w3p[j] = (row == 0) ? (_Float16)w3[col] : (_Float16)0.0f;
  }
}

// ----------------------------- main kernel ---------------------------------
// Block = 128 threads = 4 waves; one point per thread; each wave owns 32 rows
// (= two 16-row WMMA M-tiles) of the LDS staging buffers, so all s_feat/s_h
// traffic is wave-local (LDS ops are in-order within a wave).

__global__ __launch_bounds__(128)
void triplane_mlp(const float* __restrict__ coords,
                  const _Float16* __restrict__ planes,   // (3,32,32,64) f16
                  const _Float16* __restrict__ w1h,      // (128,64)  f16
                  const _Float16* __restrict__ w2h,      // (128,128) f16
                  const _Float16* __restrict__ w3p,      // (16,128)  f16, row0=w3
                  const float* __restrict__ b1,
                  const float* __restrict__ b2,
                  const float* __restrict__ b3,
                  float* __restrict__ out, int M) {
  __shared__ _Float16 s_feat[128][64];    // 16 KB : layer-1 A staging
  __shared__ _Float16 s_h[128][128];      // 32 KB : h1 then h2 (per-wave reuse)
  __shared__ float s_b1[128], s_b2[128];
  __shared__ float s_b3;

  const int tid = threadIdx.x;
  const int gbase = blockIdx.x * 128;
  const int g = gbase + tid;
  const int gc = (g < M) ? g : (M - 1);

  s_b1[tid] = b1[tid];
  s_b2[tid] = b2[tid];
  if (tid == 0) s_b3 = b3[0];

  // ---- Phase 1: triplane bilinear sampling (packed f16) -> LDS ----
  float cx = coords[3 * gc + 0];
  float cy = coords[3 * gc + 1];
  float cz = coords[3 * gc + 2];

  int axy[4], ayz[4], axz[4];
  _Float16 wxy[4], wyz[4], wxz[4];
  plane_setup(cx, cy, axy, wxy);   // xy-plane samples (x, y)
  plane_setup(cy, cz, ayz, wyz);   // yz-plane samples (y, z)
  plane_setup(cx, cz, axz, wxz);   // xz-plane samples (x, z)

  const _Float16* pxy = planes;
  const _Float16* pyz = planes + 65536;
  const _Float16* pxz = planes + 2 * 65536;

#pragma unroll
  for (int c0 = 0; c0 < 64; c0 += 16) {
    h8 a_lo, a_hi, b_lo, b_hi, c_lo, c_hi;
    sample16(pxy, axy, wxy, c0, a_lo, a_hi);
    sample16(pyz, ayz, wyz, c0, b_lo, b_hi);
    sample16(pxz, axz, wxz, c0, c_lo, c_hi);
    h8 lo = a_lo * b_lo * c_lo;      // v_pk_mul_f16
    h8 hi = a_hi * b_hi * c_hi;
    *(h8*)&s_feat[tid][c0]     = lo;
    *(h8*)&s_feat[tid][c0 + 8] = hi;
  }
  __syncthreads();

  const int lane   = tid & 31;
  const int waveId = tid >> 5;
  const int lrow   = lane & 15;
  const int sub    = lane >> 4;

  // ---- Phase 2: layer 1  h1 = relu(feat[16x64] * w1^T) via WMMA ----
#pragma unroll
  for (int mt = 0; mt < 2; ++mt) {
    const int R = waveId * 32 + mt * 16;
    const _Float16* arow = &s_feat[R + lrow][0];
    v16h a0 = load_fragA(arow, 0, sub);
    v16h a1 = load_fragA(arow, 32, sub);
#pragma unroll
    for (int t = 0; t < 8; ++t) {
      const _Float16* brow = w1h + (t * 16 + lrow) * 64;   // L2-hot global
      v16h bb0 = load_fragB(brow, 0, sub);
      v16h bb1 = load_fragB(brow, 32, sub);
      v8f acc = {};
      acc = WMMA_F16(a0, bb0, acc);
      acc = WMMA_F16(a1, bb1, acc);
      float bias = s_b1[t * 16 + lrow];
#pragma unroll
      for (int r = 0; r < 8; ++r) {
        float hv = acc[r] + bias;
        hv = fmaxf(hv, 0.0f);
        s_h[R + r + sub * 8][t * 16 + lrow] = (_Float16)hv;  // D layout scatter
      }
    }
  }
  __syncthreads();

  // ---- Phase 3+4 fused per m-tile: layer 2 (WMMA) then layer 3 (WMMA) ----
#pragma unroll
  for (int mt = 0; mt < 2; ++mt) {
    const int R = waveId * 32 + mt * 16;
    const _Float16* arow = &s_h[R + lrow][0];
    v16h a[4];
#pragma unroll
    for (int k = 0; k < 4; ++k) a[k] = load_fragA(arow, k * 32, sub);  // read h1 first
#pragma unroll
    for (int t = 0; t < 8; ++t) {
      v8f acc = {};
#pragma unroll
      for (int k = 0; k < 4; ++k) {
        const _Float16* brow = w2h + (t * 16 + lrow) * 128;
        v16h bb = load_fragB(brow, k * 32, sub);
        acc = WMMA_F16(a[k], bb, acc);
      }
      float bias = s_b2[t * 16 + lrow];
#pragma unroll
      for (int r = 0; r < 8; ++r) {
        float hv = acc[r] + bias;
        hv = fmaxf(hv, 0.0f);
        s_h[R + r + sub * 8][t * 16 + lrow] = (_Float16)hv;  // overwrite h1 with h2
      }
    }
    // Layer 3: out = h2[16x128] x w3p^T (only column N=0 of D is meaningful).
    v16h a2[4];
#pragma unroll
    for (int k = 0; k < 4; ++k) a2[k] = load_fragA(arow, k * 32, sub);  // wave-local RAW
    v8f acc3 = {};
#pragma unroll
    for (int k = 0; k < 4; ++k) {
      const _Float16* brow = w3p + lrow * 128;
      v16h bb = load_fragB(brow, k * 32, sub);
      acc3 = WMMA_F16(a2[k], bb, acc3);
    }
    if (lrow == 0) {   // lanes 0 (M=r) and 16 (M=8+r) hold column N=0
#pragma unroll
      for (int r = 0; r < 8; ++r) {
        int m = R + r + sub * 8;
        int gg = gbase + m;
        if (gg < M) out[gg] = acc3[r] + s_b3;
      }
    }
  }
}

// ----------------------------- launcher ------------------------------------

extern "C" void kernel_launch(void* const* d_in, const int* in_sizes, int n_in,
                              void* d_out, int out_size, void* d_ws, size_t ws_size,
                              hipStream_t stream) {
  const float* coords = (const float*)d_in[0];
  const float* pxy    = (const float*)d_in[1];
  const float* pyz    = (const float*)d_in[2];
  const float* pxz    = (const float*)d_in[3];
  const float* w1     = (const float*)d_in[4];
  const float* b1     = (const float*)d_in[5];
  const float* w2     = (const float*)d_in[6];
  const float* b2     = (const float*)d_in[7];
  const float* w3     = (const float*)d_in[8];
  const float* b3     = (const float*)d_in[9];
  float* out = (float*)d_out;
  const int M = in_sizes[0] / 3;

  // d_ws layout (bytes):
  //   [0, 393216)        planes f16 (3 x 32 x 32 x 64)
  //   [393216, +16384)   w1h (128x64 f16)
  //   [409600, +32768)   w2h (128x128 f16)
  //   [442368, +4096)    w3p (16x128 f16, row0 = w3, rest zero)
  _Float16* planes_h = (_Float16*)d_ws;
  _Float16* w1h = (_Float16*)((char*)d_ws + 393216);
  _Float16* w2h = (_Float16*)((char*)d_ws + 409600);
  _Float16* w3p = (_Float16*)((char*)d_ws + 442368);

  prep_planes <<<768, 256, 0, stream>>>(pxy, pyz, pxz, planes_h);
  prep_weights<<<104, 256, 0, stream>>>(w1, w2, w3, w1h, w2h, w3p);

  const int grid = (M + 127) / 128;
  triplane_mlp<<<grid, 128, 0, stream>>>(coords, planes_h, w1h, w2h, w3p,
                                         b1, b2, b3, out, M);
}
